// ParallelExperts_88287347736702
// MI455X (gfx1250) — matrix operand order repaired and checked by reading, not compiled, not run on hardware
//
#include <hip/hip_runtime.h>
#include <hip/hip_bf16.h>

// ---------------------------------------------------------------------------
// MoE grouped GEMM for MI455X (gfx1250, wave32, WMMA).
// f32 reference computed via bf16-split WMMA: a = a_hi + a_lo (bf16 each),
// acc += A_hi*B_hi + A_hi*B_lo + A_lo*B_hi  (f32 accumulate).
// TILE_M=64: one B hi/lo fragment feeds 12 WMMAs (4 A tiles x 3 terms),
// keeping the loop matrix-pipe bound instead of conversion-VALU bound.
// ---------------------------------------------------------------------------

typedef __attribute__((ext_vector_type(16))) __bf16 v16bf;
typedef __attribute__((ext_vector_type(8)))  float  v8f;

constexpr int N_TOK  = 2048;
constexpr int D_IN   = 2048;
constexpr int D_OUT  = 2048;
constexpr int NEXP   = 8;
constexpr int TOPK_C = 2;
constexpr int NKROWS = N_TOK * TOPK_C;           // 4096 flat (token,slot) rows

constexpr int TILE_M = 64;                       // rows per block (4 WMMA tiles)
constexpr int TILE_N = 64;                       // cols per block (4 waves x 16)
constexpr int TILE_K = 32;                       // K per WMMA step (bf16)
constexpr int LDS_STRIDE = 40;                   // padded row stride in shorts (80B)
constexpr int MAX_ROW_TILES = NKROWS / TILE_M + NEXP;  // 72 upper bound

// Native conversions: fptrunc f32->bf16 is round-to-nearest-even.
__device__ __forceinline__ void split_bf16(float f, __bf16& hi, __bf16& lo) {
  hi = (__bf16)f;
  lo = (__bf16)(f - (float)hi);
}

union Frag { v16bf v; __bf16 h[16]; uint4 q[2]; };
union StageU { __bf16 h[16]; uint4 q[2]; };

__global__ void moe_zero_kernel(float* __restrict__ out, int n) {
  int i = (blockIdx.x * blockDim.x + threadIdx.x) * 4;
  if (i + 3 < n) {
    *(float4*)(out + i) = make_float4(0.f, 0.f, 0.f, 0.f);
  } else {
    for (int j = i; j < n; ++j) out[j] = 0.f;
  }
}

__global__ __launch_bounds__(128) void moe_wmma_kernel(
    const float* __restrict__ inputs,            // [N, D_IN]
    const float* __restrict__ weight,            // [E, D_OUT, D_IN]
    const int*   __restrict__ kptr,              // scalar TOPK
    const int*   __restrict__ sorted_scattered,  // [N*k] flat idx per sorted pos
    const int*   __restrict__ expert_offsets,    // [E] inclusive cumsum
    const float* __restrict__ gates,             // [N, k]
    float*       __restrict__ out)               // [N, D_OUT]
{
  __shared__ __bf16 ldsAhi[TILE_M * LDS_STRIDE];
  __shared__ __bf16 ldsAlo[TILE_M * LDS_STRIDE];
  __shared__ int    tokLds[TILE_M];
  __shared__ float  gateLds[TILE_M];

  // Map blockIdx.x -> (expert, 64-row tile inside that expert's group).
  int bid = blockIdx.x;
  int e = 0, rowStart = 0, rowEnd = 0, prev = 0;
  bool found = false;
  for (e = 0; e < NEXP; ++e) {
    int end = expert_offsets[e];
    int cnt = end - prev;
    int nt  = (cnt + TILE_M - 1) / TILE_M;
    if (bid < nt) { rowStart = prev + bid * TILE_M; rowEnd = end; found = true; break; }
    bid -= nt; prev = end;
  }
  if (!found) return;                            // uniform early exit

  const int tid  = threadIdx.x;
  const int wave = tid >> 5;
  const int lane = tid & 31;
  const int kk   = kptr[0];

  // Per-row token + gate (gate = 0 masks rows past the expert group's end).
  if (tid < TILE_M) {
    int j = rowStart + tid;
    int tok = 0; float g = 0.f;
    if (j < rowEnd) {
      int flat = sorted_scattered[j];
      tok = flat / kk;
      int slot = flat - tok * kk;
      g = gates[tok * kk + slot];
    }
    tokLds[tid]  = tok;
    gateLds[tid] = g;
  }
  __syncthreads();

  // B: lane owns column n; lanes 0-15 carry K 0..15 of each 32-chunk,
  // lanes 16-31 carry K 16..31 (contiguous k order, same as A).
  const int nc   = blockIdx.y * TILE_N + wave * 16 + (lane & 15);
  const int koff = (lane < 16) ? 0 : 16;
  const float* __restrict__ wptr =
      weight + ((size_t)e * D_OUT + (size_t)nc) * D_IN + koff;

  // A staging: 2 threads per row, 16 consecutive f32 each.
  const int ar = tid >> 1;                       // 0..63 row in tile
  const int ac = (tid & 1) * 16;                 // 0 or 16 col chunk
  const float* __restrict__ aptr = inputs + (size_t)tokLds[ar] * D_IN + ac;

  // A fragment geometry (contiguous k, same order as B): lane row-in-tile m = lane&15,
  // chunk offsets {o, o+8} with o = 0 (lanes 0-15) or 16 (lanes 16-31).
  const int am = lane & 15;
  const int ao = (lane < 16) ? 0 : 16;

  v8f acc[4] = {{}, {}, {}, {}};                 // 4 x (16 rows) of the M tile

  for (int kb = 0; kb < D_IN; kb += TILE_K) {
    // ---- per-lane B load (global f32) + hi/lo bf16 split (registers) ----
    Frag bhi, blo;
    const float* wk = wptr + kb;
#pragma unroll
    for (int h4 = 0; h4 < 4; ++h4) {
      float4 w = *(const float4*)(wk + h4 * 4);
      float wv[4] = {w.x, w.y, w.z, w.w};
#pragma unroll
      for (int i = 0; i < 4; ++i)
        split_bf16(wv[i], bhi.h[h4 * 4 + i], blo.h[h4 * 4 + i]);
    }
    if (kb + TILE_K < D_IN)
      __builtin_prefetch((const void*)(wk + TILE_K), 0, 1);  // global_prefetch_b8

    // ---- cooperative A stage into LDS (hi/lo bf16, padded rows) ----
    {
      StageU shi, slo;
#pragma unroll
      for (int h4 = 0; h4 < 4; ++h4) {
        float4 a = *(const float4*)(aptr + kb + h4 * 4);
        float av[4] = {a.x, a.y, a.z, a.w};
#pragma unroll
        for (int i = 0; i < 4; ++i)
          split_bf16(av[i], shi.h[h4 * 4 + i], slo.h[h4 * 4 + i]);
      }
      uint4* dhi = (uint4*)(ldsAhi + ar * LDS_STRIDE + ac);
      uint4* dlo = (uint4*)(ldsAlo + ar * LDS_STRIDE + ac);
      dhi[0] = shi.q[0]; dhi[1] = shi.q[1];      // ds_store_b128 x2
      dlo[0] = slo.q[0]; dlo[1] = slo.q[1];
    }
    __syncthreads();

    // ---- 4 A tiles x 3 split terms = 12 WMMAs per staged chunk ----
#pragma unroll
    for (int tt = 0; tt < 4; ++tt) {
      const __bf16* hbase = ldsAhi + (am + 16 * tt) * LDS_STRIDE;
      const __bf16* lbase = ldsAlo + (am + 16 * tt) * LDS_STRIDE;
      Frag ahi, alo;
      ahi.q[0] = *(const uint4*)(hbase + ao);
      ahi.q[1] = *(const uint4*)(hbase + ao + 8);
      alo.q[0] = *(const uint4*)(lbase + ao);
      alo.q[1] = *(const uint4*)(lbase + ao + 8);
      acc[tt] = __builtin_amdgcn_wmma_f32_16x16x32_bf16(false, ahi.v, false, bhi.v,
                                                        (short)0, acc[tt], false, false);
      acc[tt] = __builtin_amdgcn_wmma_f32_16x16x32_bf16(false, ahi.v, false, blo.v,
                                                        (short)0, acc[tt], false, false);
      acc[tt] = __builtin_amdgcn_wmma_f32_16x16x32_bf16(false, alo.v, false, bhi.v,
                                                        (short)0, acc[tt], false, false);
    }
    __syncthreads();                             // protect LDS reuse next iter
  }

  // ---- epilogue: C/D layout VGPR r -> M = r (+8 for hi lanes); lane -> N.
  // Scale by gate and scatter-add into out[token] (TOPK contributions sum).
#pragma unroll
  for (int tt = 0; tt < 4; ++tt) {
#pragma unroll
    for (int r = 0; r < 8; ++r) {
      int m = 16 * tt + r + ((lane < 16) ? 0 : 8);
      float v = acc[tt][r] * gateLds[m];
      unsafeAtomicAdd(&out[(size_t)tokLds[m] * D_OUT + nc], v);  // global_atomic_add_f32
    }
  }
}

extern "C" void kernel_launch(void* const* d_in, const int* in_sizes, int n_in,
                              void* d_out, int out_size, void* d_ws, size_t ws_size,
                              hipStream_t stream) {
  const float* inputs           = (const float*)d_in[0];
  const float* weight           = (const float*)d_in[1];
  const int*   kptr             = (const int*)d_in[2];
  // d_in[3] = sorted_expert_idxs (implied by expert_offsets grouping; unused)
  const int*   sorted_scattered = (const int*)d_in[4];
  // d_in[5] = padded_block_idxs (no math effect)
  const int*   expert_offsets   = (const int*)d_in[6];
  const float* gates            = (const float*)d_in[7];
  float*       out              = (float*)d_out;

  // Zero the output accumulator every call (atomic-scatter epilogue).
  int vec4 = (out_size + 3) / 4;
  int zb   = 256;
  moe_zero_kernel<<<(vec4 + zb - 1) / zb, zb, 0, stream>>>(out, out_size);

  dim3 grid(MAX_ROW_TILES, D_OUT / TILE_N);      // 72 x 32
  moe_wmma_kernel<<<grid, 128, 0, stream>>>(
      inputs, weight, kptr, sorted_scattered, expert_offsets, gates, out);
}
